// CausalSelfAttention_89919435309097
// MI455X (gfx1250) — matrix operand order repaired and checked by reading, not compiled, not run on hardware
//
#include <hip/hip_runtime.h>
#include <hip/hip_bf16.h>

// ---------------------------------------------------------------------------
// Problem constants (match reference)
// ---------------------------------------------------------------------------
#define BB 2
#define TT 2048
#define CC 2048
#define HH_Q 16
#define HH_KV 4
#define HD 128
#define NREP 4

typedef __attribute__((ext_vector_type(16))) __bf16 v16bf;
typedef __attribute__((ext_vector_type(8)))  __bf16 v8bf;
typedef __attribute__((ext_vector_type(8)))  float  v8f;

static __device__ __forceinline__ float bf2f(__bf16 h) {
    unsigned short u = __builtin_bit_cast(unsigned short, h);
    unsigned x = ((unsigned)u) << 16;
    return __builtin_bit_cast(float, x);
}
static __device__ __forceinline__ __bf16 f2bf(float f) {
    unsigned x = __builtin_bit_cast(unsigned, f);
    unsigned r = x + 0x7FFFu + ((x >> 16) & 1u);   // RNE
    return __builtin_bit_cast(__bf16, (unsigned short)(r >> 16));
}
static __device__ __forceinline__ v16bf cat8(v8bf lo, v8bf hi) {
    return __builtin_shufflevector(lo, hi, 0,1,2,3,4,5,6,7,8,9,10,11,12,13,14,15);
}

// ---------------------------------------------------------------------------
// gfx1250 async copy: global -> LDS, 16 bytes, tracked by ASYNCcnt.
// LDS byte address = low 32 bits of the generic pointer (aperture truncation).
// ---------------------------------------------------------------------------
static __device__ __forceinline__ void async_cp16(void* lds, const void* gptr) {
    unsigned l = (unsigned)(size_t)lds;
    unsigned long long g = (unsigned long long)(size_t)gptr;
    asm volatile("global_load_async_to_lds_b128 %0, %1, off"
                 :: "v"(l), "v"(g) : "memory");
}
static __device__ __forceinline__ void wait_async_le4() {
    asm volatile("s_wait_asynccnt 4" ::: "memory");
}
static __device__ __forceinline__ void wait_async_0() {
    asm volatile("s_wait_asynccnt 0" ::: "memory");
}
// wave-local LDS ordering (same-wave DS ops execute in order; this stops the
// compiler from reordering around the store->load sequence)
static __device__ __forceinline__ void wave_fence() {
    asm volatile("" ::: "memory");
    __builtin_amdgcn_wave_barrier();
}

// ---------------------------------------------------------------------------
// Tensor Data Mover: 2D tile load global -> LDS (D# per CDNA5 ISA ch.8).
// Units of dims/strides: elements (data_size = 2 bytes).
// ---------------------------------------------------------------------------
typedef __attribute__((ext_vector_type(4))) unsigned tdm_v4u;
typedef __attribute__((ext_vector_type(8))) int      tdm_v8i;
typedef __attribute__((ext_vector_type(4))) int      tdm_v4i;

static __device__ __forceinline__ void tdm_load_2d(
    unsigned lds_addr, const void* gptr,
    unsigned tensor_d0, unsigned tensor_d1,
    unsigned tile_d0, unsigned tile_d1, unsigned stride0)
{
    unsigned long long ga = (unsigned long long)(size_t)gptr;
    tdm_v4u g0;
    g0[0] = 1u;                                          // count=1, user D#
    g0[1] = lds_addr;                                    // lds_addr [63:32]
    g0[2] = (unsigned)(ga & 0xFFFFFFFFu);                // global_addr[31:0]
    g0[3] = (unsigned)((ga >> 32) & 0x1FFFFFFu) | (2u << 30);  // addr[56:32] | type=2
    tdm_v8i g1;
    g1[0] = (int)(1u << 16);                             // data_size=1 (2B)
    g1[1] = (int)((tensor_d0 & 0xFFFFu) << 16);          // tensor_dim0 lo16 @48
    g1[2] = (int)(((tensor_d0 >> 16) & 0xFFFFu) | ((tensor_d1 & 0xFFFFu) << 16));
    g1[3] = (int)(((tensor_d1 >> 16) & 0xFFFFu) | ((tile_d0 & 0xFFFFu) << 16));
    g1[4] = (int)(tile_d1 & 0xFFFFu);                    // tile_dim1; tile_dim2=0
    g1[5] = (int)stride0;                                // tensor_dim0_stride lo32
    g1[6] = 0;
    g1[7] = 0;
    tdm_v4i z4 = {};
#if __clang_major__ >= 23
    tdm_v8i z8 = {};
    __builtin_amdgcn_tensor_load_to_lds(g0, g1, z4, z4, z8, 0);
#else
    __builtin_amdgcn_tensor_load_to_lds(g0, g1, z4, z4, 0);
#endif
}

// ---------------------------------------------------------------------------
// fp32 -> bf16 elementwise convert
// ---------------------------------------------------------------------------
__global__ void k_cvt_bf16(const float* __restrict__ in, __bf16* __restrict__ out, long n) {
    long i = (long)blockIdx.x * blockDim.x + threadIdx.x;
    if (i < n) out[i] = f2bf(in[i]);
}

// fp32 (K,N) -> bf16 (N,K) transpose-convert
__global__ void k_transpose_w(const float* __restrict__ in, __bf16* __restrict__ out, int K, int N) {
    long i = (long)blockIdx.x * blockDim.x + threadIdx.x;
    long tot = (long)K * N;
    if (i >= tot) return;
    int k = (int)(i / N);
    int n = (int)(i % N);
    out[(long)n * K + k] = f2bf(in[i]);
}

// ---------------------------------------------------------------------------
// bf16 WMMA GEMM, LDS-tiled + async double buffering (last K-step peeled so
// the steady-state loop has no branches):
//   out(M,N) = A(M,K) @ Bt(N,K)^T + bias
// Block: 256 threads = 8 waves, 128x128 tile; wave computes 32x64.
// ---------------------------------------------------------------------------
__global__ __launch_bounds__(256) void k_gemm_bf16(
    const __bf16* __restrict__ A, const __bf16* __restrict__ Bt,
    const float* __restrict__ bias,
    __bf16* __restrict__ outb, float* __restrict__ outf,
    int M, int N, int K)
{
    __shared__ __align__(16) __bf16 sA[2][128][32];
    __shared__ __align__(16) __bf16 sB[2][128][32];

    const int lane = threadIdx.x;
    const int w    = threadIdx.y;                 // wave 0..7
    const int tid  = w * 32 + lane;
    const int mrow = lane & 15;
    const int hh   = lane >> 4;
    const int wr   = w >> 1;                      // 0..3 -> m offset
    const int wc   = w & 1;                       // 0..1 -> n offset
    const int mb   = blockIdx.x * 128;
    const int nb   = blockIdx.y * 128;

    v8f acc[2][4];
    #pragma unroll
    for (int i = 0; i < 2; ++i)
        #pragma unroll
        for (int j = 0; j < 4; ++j) acc[i][j] = v8f{};

    auto issue_tile = [&](int buf, int kk) {
        #pragma unroll
        for (int s = 0; s < 2; ++s) {
            int seg = tid + s * 256;              // 0..511
            int r = seg >> 1, c = seg & 1;        // row, 16B half of 64B row
            async_cp16(&sA[buf][r][c * 16], A  + (size_t)(mb + r) * K + kk + c * 16);
            async_cp16(&sB[buf][r][c * 16], Bt + (size_t)(nb + r) * K + kk + c * 16);
        }
    };

    auto compute = [&](int buf) {
        v16bf af[2];
        #pragma unroll
        for (int i = 0; i < 2; ++i) {
            int ml = wr * 32 + i * 16 + mrow;
            af[i] = *(const v16bf*)&sA[buf][ml][hh * 16];
        }
        #pragma unroll
        for (int j = 0; j < 4; ++j) {
            int nl = wc * 64 + j * 16 + mrow;
            v16bf bf = *(const v16bf*)&sB[buf][nl][hh * 16];
            #pragma unroll
            for (int i = 0; i < 2; ++i)
                acc[i][j] = __builtin_amdgcn_wmma_f32_16x16x32_bf16(
                    false, af[i], false, bf, (short)0, acc[i][j], false, false);
        }
    };

    issue_tile(0, 0);
    int it = 0;
    for (int kk = 0; kk < K - 32; kk += 32, ++it) {   // steady state: no branches
        const int buf = it & 1;
        issue_tile(buf ^ 1, kk + 32);
        wait_async_le4();
        __syncthreads();
        compute(buf);
        __syncthreads();
    }
    wait_async_0();                                    // peeled last K-step
    __syncthreads();
    compute(it & 1);

    #pragma unroll
    for (int i = 0; i < 2; ++i) {
        #pragma unroll
        for (int j = 0; j < 4; ++j) {
            int n = nb + wc * 64 + j * 16 + mrow;
            float bv = bias ? bias[n] : 0.0f;
            #pragma unroll
            for (int r = 0; r < 8; ++r) {
                int row = mb + wr * 32 + i * 16 + r + 8 * hh;
                float v = acc[i][j][r] + bv;
                if (outf) outf[(size_t)row * N + n] = v;
                else      outb[(size_t)row * N + n] = f2bf(v);
            }
        }
    }
}

// ---------------------------------------------------------------------------
// RoPE on Q + relayout (b,t,h,d) -> (b,h,t,d)
// ---------------------------------------------------------------------------
__global__ void k_rope_q(const __bf16* __restrict__ qlin,
                         const float* __restrict__ cosb, const float* __restrict__ sinb,
                         __bf16* __restrict__ qh)
{
    long i = (long)blockIdx.x * blockDim.x + threadIdx.x;
    if (i >= (long)BB * TT * CC) return;
    int b = (int)(i / ((long)TT * CC));
    int rem = (int)(i % ((long)TT * CC));
    int t = rem / CC;
    int c = rem % CC;
    int h = c >> 7;
    int d = c & 127;
    float qv = bf2f(qlin[i]);
    float rv = (d < 64) ? -bf2f(qlin[i + 64]) : bf2f(qlin[i - 64]);
    float cv = cosb[((size_t)b * TT + t) * HD + d];
    float sv = sinb[((size_t)b * TT + t) * HD + d];
    qh[(((size_t)b * HH_Q + h) * TT + t) * HD + d] = f2bf(qv * cv + rv * sv);
}

// RoPE on K -> (b,hk,t,d); V copy -> transposed (b,hk,d,t)
__global__ void k_rope_kv(const __bf16* __restrict__ klin, const __bf16* __restrict__ vlin,
                          const float* __restrict__ cosb, const float* __restrict__ sinb,
                          __bf16* __restrict__ kh, __bf16* __restrict__ vT)
{
    long i = (long)blockIdx.x * blockDim.x + threadIdx.x;
    const int CKV = HH_KV * HD;  // 512
    if (i >= (long)BB * TT * CKV) return;
    int b = (int)(i / ((long)TT * CKV));
    int rem = (int)(i % ((long)TT * CKV));
    int t = rem / CKV;
    int c = rem % CKV;
    int hk = c >> 7;
    int d  = c & 127;
    float kv = bf2f(klin[i]);
    float rv = (d < 64) ? -bf2f(klin[i + 64]) : bf2f(klin[i - 64]);
    float cv = cosb[((size_t)b * TT + t) * HD + d];
    float sv = sinb[((size_t)b * TT + t) * HD + d];
    kh[(((size_t)b * HH_KV + hk) * TT + t) * HD + d] = f2bf(kv * cv + rv * sv);
    vT[(((size_t)b * HH_KV + hk) * HD + d) * TT + t] = vlin[i];
}

// ---------------------------------------------------------------------------
// Flash attention: block = 4 waves = 4 consecutive 16-row q tiles of one
// (b,h).  K/V chunks (32 kv) staged in LDS by the Tensor Data Mover (one
// wave issues 2 D# loads, waits TENSORcnt, barrier publishes to all waves).
// Causal online softmax; P transposed via per-wave LDS scratch.
// ---------------------------------------------------------------------------
__global__ __launch_bounds__(128) void k_flash(
    const __bf16* __restrict__ qh, const __bf16* __restrict__ kh,
    const __bf16* __restrict__ vT, __bf16* __restrict__ y)
{
    __shared__ __align__(16) __bf16 sK[32][128];    // kv row x hd   (8 KB tile)
    __shared__ __align__(16) __bf16 sV[128][32];    // hd row x kv   (8 KB tile)
    __shared__ __align__(16) __bf16 pl[4][16][32];  // per-wave P scratch

    const int lane = threadIdx.x;
    const int w    = threadIdx.y;                   // wave 0..3
    const int n    = lane & 15;
    const int hh   = lane >> 4;
    const int QT   = TT / 16;

    int id  = blockIdx.x;                           // over (b, h, qblock)
    int qb  = id % (QT / 4);
    int h   = (id / (QT / 4)) % HH_Q;
    int b   = id / ((QT / 4) * HH_Q);
    int hk  = h / NREP;
    int qt  = qb * 4 + w;                           // this wave's q tile

    // Q tile (16 x 128) as 4 A-fragments
    const __bf16* qbase = qh + (((size_t)b * HH_Q + h) * TT + qt * 16 + n) * HD;
    v16bf qa[4];
    #pragma unroll
    for (int c = 0; c < 4; ++c) {
        qa[c] = *(const v16bf*)(qbase + c * 32 + hh * 16);
    }

    v8f o[8];
    #pragma unroll
    for (int nt = 0; nt < 8; ++nt) o[nt] = v8f{};
    float mrow[8], lrow[8];
    #pragma unroll
    for (int r = 0; r < 8; ++r) { mrow[r] = -1e30f; lrow[r] = 0.0f; }

    const __bf16* kbase = kh + ((size_t)b * HH_KV + hk) * (size_t)TT * HD;
    const __bf16* vbase = vT + ((size_t)b * HH_KV + hk) * (size_t)HD * TT;
    const float scale = 0.08838834764831845f;       // 1/sqrt(128)

    const int nch_me  = (qt + 2) >> 1;              // chunks this wave needs
    const int nch_max = (qb * 4 + 3 + 2) >> 1;      // chunks block must load
    const unsigned sK_lds = (unsigned)(size_t)&sK[0][0];
    const unsigned sV_lds = (unsigned)(size_t)&sV[0][0];

    for (int ch = 0; ch < nch_max; ++ch) {
        __syncthreads();                            // prior chunk reads done
        if (w == 0) {
            // K tile: rows ch*32..+31, full 128-wide rows, row stride 128
            tdm_load_2d(sK_lds, kbase + (size_t)(ch * 32) * HD,
                        HD, TT, HD, 32, HD);
            // V tile: 128 rows x 32 cols starting at column ch*32, stride TT
            tdm_load_2d(sV_lds, vbase + ch * 32,
                        TT, HD, 32, HD, TT);
            __builtin_amdgcn_s_wait_tensorcnt(0);
        }
        __syncthreads();                            // tiles visible to all waves

        if (ch < nch_me) {
            v8f s2[2];
            #pragma unroll
            for (int t = 0; t < 2; ++t) {
                v8f acc = v8f{};
                #pragma unroll
                for (int c = 0; c < 4; ++c) {
                    v16bf kf = *(const v16bf*)&sK[t * 16 + n][c * 32 + hh * 16];
                    acc = __builtin_amdgcn_wmma_f32_16x16x32_bf16(
                        false, qa[c], false, kf, (short)0, acc, false, false);
                }
                #pragma unroll
                for (int r = 0; r < 8; ++r) {
                    int row = qt * 16 + r + 8 * hh;
                    int col = ch * 32 + t * 16 + n;
                    float sv = acc[r] * scale;
                    if (col > row) sv = -1e30f;
                    acc[r] = sv;
                }
                s2[t] = acc;
            }

            float p0v[8], p1v[8];
            #pragma unroll
            for (int r = 0; r < 8; ++r) {
                float mx = fmaxf(s2[0][r], s2[1][r]);
                #pragma unroll
                for (int d = 1; d < 16; d <<= 1) mx = fmaxf(mx, __shfl_xor(mx, d, 32));
                float mn = fmaxf(mrow[r], mx);
                float alpha = __expf(mrow[r] - mn);
                float p0 = __expf(s2[0][r] - mn);
                float p1 = __expf(s2[1][r] - mn);
                float rs = p0 + p1;
                #pragma unroll
                for (int d = 1; d < 16; d <<= 1) rs += __shfl_xor(rs, d, 32);
                lrow[r] = lrow[r] * alpha + rs;
                mrow[r] = mn;
                p0v[r] = p0; p1v[r] = p1;
                #pragma unroll
                for (int nt = 0; nt < 8; ++nt) o[nt][r] *= alpha;
            }

            // transpose P (D layout) -> A layout via per-wave LDS scratch
            wave_fence();
            #pragma unroll
            for (int r = 0; r < 8; ++r) {
                pl[w][r + 8 * hh][n]      = f2bf(p0v[r]);
                pl[w][r + 8 * hh][16 + n] = f2bf(p1v[r]);
            }
            wave_fence();
            v16bf pa = *(const v16bf*)&pl[w][n][hh * 16];

            // O += P(16x32) @ V(32x128)
            #pragma unroll
            for (int nt = 0; nt < 8; ++nt) {
                v16bf vf = *(const v16bf*)&sV[nt * 16 + n][hh * 16];
                o[nt] = __builtin_amdgcn_wmma_f32_16x16x32_bf16(
                    false, pa, false, vf, (short)0, o[nt], false, false);
            }
        }
    }

    // normalize + store to (b,t,c) bf16
    #pragma unroll
    for (int r = 0; r < 8; ++r) {
        float inv = 1.0f / lrow[r];
        int row = qt * 16 + r + 8 * hh;
        #pragma unroll
        for (int nt = 0; nt < 8; ++nt) {
            y[((size_t)b * TT + row) * CC + h * HD + nt * 16 + n] = f2bf(o[nt][r] * inv);
        }
    }
}

// ---------------------------------------------------------------------------
// Host-side launch
// ---------------------------------------------------------------------------
extern "C" void kernel_launch(void* const* d_in, const int* in_sizes, int n_in,
                              void* d_out, int out_size, void* d_ws, size_t ws_size,
                              hipStream_t stream) {
    (void)in_sizes; (void)n_in; (void)out_size; (void)ws_size;
    const float* x    = (const float*)d_in[0];
    const float* cosb = (const float*)d_in[1];
    const float* sinb = (const float*)d_in[2];
    const float* Wq   = (const float*)d_in[3];
    const float* bq   = (const float*)d_in[4];
    const float* Wk   = (const float*)d_in[5];
    const float* bk   = (const float*)d_in[6];
    const float* Wv   = (const float*)d_in[7];
    const float* bv   = (const float*)d_in[8];
    const float* Wo   = (const float*)d_in[9];
    float* out = (float*)d_out;

    const size_t M = (size_t)BB * TT;          // 4096
    char* w = (char*)d_ws;
    size_t off = 0;
    auto take = [&](size_t bytes) { char* p = w + off; off += (bytes + 255) & ~(size_t)255; return p; };

    __bf16* xb   = (__bf16*)take(M * CC * 2);
    __bf16* WqT  = (__bf16*)take((size_t)CC * CC * 2);
    __bf16* WkT  = (__bf16*)take((size_t)512 * CC * 2);
    __bf16* WvT  = (__bf16*)take((size_t)512 * CC * 2);
    __bf16* WoT  = (__bf16*)take((size_t)CC * CC * 2);
    __bf16* qlin = (__bf16*)take(M * CC * 2);
    __bf16* klin = (__bf16*)take(M * 512 * 2);
    __bf16* vlin = (__bf16*)take(M * 512 * 2);
    __bf16* qhb  = (__bf16*)take(M * CC * 2);
    __bf16* khb  = (__bf16*)take(M * 512 * 2);
    __bf16* vTb  = (__bf16*)take(M * 512 * 2);
    __bf16* ybuf = qlin;   // q-linear dead after RoPE; reuse as attention output

    // 1) convert / transpose to bf16
    {
        long n = (long)M * CC;
        k_cvt_bf16<<<dim3((unsigned)((n + 255) / 256)), dim3(256), 0, stream>>>(x, xb, n);
    }
    k_transpose_w<<<dim3((unsigned)(((size_t)CC * CC + 255) / 256)), dim3(256), 0, stream>>>(Wq, WqT, CC, CC);
    k_transpose_w<<<dim3((unsigned)(((size_t)CC * 512 + 255) / 256)), dim3(256), 0, stream>>>(Wk, WkT, CC, 512);
    k_transpose_w<<<dim3((unsigned)(((size_t)CC * 512 + 255) / 256)), dim3(256), 0, stream>>>(Wv, WvT, CC, 512);
    k_transpose_w<<<dim3((unsigned)(((size_t)CC * CC + 255) / 256)), dim3(256), 0, stream>>>(Wo, WoT, CC, CC);

    // 2) QKV projections (WMMA, async LDS-tiled)
    k_gemm_bf16<<<dim3((unsigned)(M / 128), CC / 128), dim3(32, 8), 0, stream>>>(
        xb, WqT, bq, qlin, nullptr, (int)M, CC, CC);
    k_gemm_bf16<<<dim3((unsigned)(M / 128), 512 / 128), dim3(32, 8), 0, stream>>>(
        xb, WkT, bk, klin, nullptr, (int)M, 512, CC);
    k_gemm_bf16<<<dim3((unsigned)(M / 128), 512 / 128), dim3(32, 8), 0, stream>>>(
        xb, WvT, bv, vlin, nullptr, (int)M, 512, CC);

    // 3) RoPE + relayout
    {
        long n = (long)M * CC;
        k_rope_q<<<dim3((unsigned)((n + 255) / 256)), dim3(256), 0, stream>>>(qlin, cosb, sinb, qhb);
        long n2 = (long)M * 512;
        k_rope_kv<<<dim3((unsigned)((n2 + 255) / 256)), dim3(256), 0, stream>>>(
            klin, vlin, cosb, sinb, khb, vTb);
    }

    // 4) flash attention (WMMA + TDM K/V staging)
    k_flash<<<dim3(BB * HH_Q * (TT / 16 / 4)), dim3(32, 4), 0, stream>>>(qhb, khb, vTb, ybuf);

    // 5) output projection (WMMA) -> fp32
    k_gemm_bf16<<<dim3((unsigned)(M / 128), CC / 128), dim3(32, 8), 0, stream>>>(
        ybuf, WoT, nullptr, nullptr, out, (int)M, CC, CC);
}
